// SelfAttention_21234318312242
// MI455X (gfx1250) — matrix operand order repaired and checked by reading, not compiled, not run on hardware
//
#include <hip/hip_runtime.h>

// ---------------------------------------------------------------------------
// Causal self-attention for MI455X (gfx1250), wave32 + WMMA f16->f32 +
// async global->LDS staging (ASYNCcnt) with double-buffered tiles.
//   x:     [4, 2048, 1024] f32
//   W_qkv: [3072, 1024]    f32
//   out:   [4, 16, 2048, 64] f32
// Stage 0: convert x, W to f16 (one-time, memory bound).
// Stage 1: qkv = x @ W^T via v_wmma_f32_16x16x32_f16, async-LDS staged,
//          writes Q (pre-scaled), K as [b,h,t,64] f16, V^T as [b,h,d,t] f16.
// Stage 2: flash attention: S=QK^T (WMMA), online softmax, O+=PV (WMMA).
// ---------------------------------------------------------------------------

typedef __attribute__((ext_vector_type(16))) _Float16 v16h;
typedef __attribute__((ext_vector_type(8)))  _Float16 v8h;
typedef __attribute__((ext_vector_type(4)))  _Float16 v4h;
typedef __attribute__((ext_vector_type(8)))  float    v8f;

union V16 { v16h v; struct { v8h lo; v8h hi; } p; };

__device__ __forceinline__ v8f wmma_f16(v16h a, v16h b, v8f c) {
    return __builtin_amdgcn_wmma_f32_16x16x32_f16(false, a, false, b,
                                                  (short)0, c, false, false);
}

// Async DMA of 16 bytes per lane: global -> LDS, tracked by ASYNCcnt.
// VDST holds the LDS byte address (low 32 bits of the generic LDS address).
__device__ __forceinline__ void async_b128(void* lds_dst, const void* gsrc) {
    uint32_t l = (uint32_t)(uintptr_t)lds_dst;
    uint64_t g = (uint64_t)(uintptr_t)gsrc;
    asm volatile("global_load_async_to_lds_b128 %0, %1, off"
                 :: "v"(l), "v"(g) : "memory");
}
#define WAIT_ASYNC(n) asm volatile("s_wait_asynccnt " #n ::: "memory")

constexpr int D_MODEL = 1024;
constexpr int NHEAD   = 16;
constexpr int SEQ     = 2048;
constexpr int BATCH   = 4;
constexpr int HD      = 64;

// ---------------------------------------------------------------------------
// Kernel 0: f32 -> f16 conversion (x and W), float4 -> packed half4.
// ---------------------------------------------------------------------------
__global__ __launch_bounds__(256) void cvt_f16_kernel(
    const float* __restrict__ src, _Float16* __restrict__ dst, int n4)
{
    int i = blockIdx.x * 256 + threadIdx.x;
    if (i < n4) {
        float4 f = ((const float4*)src)[i];
        v4h h = { (_Float16)f.x, (_Float16)f.y, (_Float16)f.z, (_Float16)f.w };
        ((v4h*)dst)[i] = h;
    }
}

// ---------------------------------------------------------------------------
// Kernel 1: fused QKV projection GEMM (f16 in, f32 acc).
//   M = 8192, N = 3072, K = 1024. Tile 128x64x32, 8 waves (4x2), 32x32/wave.
//   A/B tiles staged with global_load_async_to_lds_b128, double buffered.
// ---------------------------------------------------------------------------
constexpr int GBM = 128, GBN = 64, GBK = 32;
constexpr int LDA = GBK + 8;   // 80B rows: every 16B chunk stays 16B-aligned
constexpr int LDB = GBK + 8;
constexpr int NKT = D_MODEL / GBK;   // 32 k-tiles

__global__ __launch_bounds__(256) void qkv_gemm_kernel(
    const _Float16* __restrict__ xh, const _Float16* __restrict__ wh,
    _Float16* __restrict__ qbuf, _Float16* __restrict__ kbuf,
    _Float16* __restrict__ vtbuf)
{
    __shared__ _Float16 sA[2][GBM][LDA];
    __shared__ _Float16 sB[2][GBN][LDB];

    const int m0  = blockIdx.x * GBM;
    const int f0  = blockIdx.y * GBN;
    const int tid = threadIdx.x;
    const int lane = tid & 31;
    const int wave = tid >> 5;
    const int wm = wave >> 1;                  // 0..3
    const int wn = wave & 1;                   // 0..1
    const int g  = lane >> 4;
    const int lm = lane & 15;

    // 3 async b128 ops per thread per k-tile (A: 2, B: 1)
    auto issue_tile = [&](int kt, int buf) {
        const int kb = kt * GBK;
        #pragma unroll
        for (int j = 0; j < 2; ++j) {
            int idx = tid + j * 256;           // 512 chunks: 128 rows x 4
            int r   = idx >> 2;
            int c   = (idx & 3) * 8;           // halves
            async_b128(&sA[buf][r][c], xh + (size_t)(m0 + r) * D_MODEL + kb + c);
        }
        {
            int r = tid >> 2;                  // 256 chunks: 64 rows x 4
            int c = (tid & 3) * 8;
            async_b128(&sB[buf][r][c], wh + (size_t)(f0 + r) * D_MODEL + kb + c);
        }
    };

    v8f acc[2][2] = {};

    issue_tile(0, 0);
    for (int kt = 0; kt < NKT; ++kt) {
        const int cur = kt & 1;
        if (kt + 1 < NKT) {
            issue_tile(kt + 1, cur ^ 1);       // prefetch next tile
            WAIT_ASYNC(0x3);                   // oldest 3 (= tile kt) retired
        } else {
            WAIT_ASYNC(0x0);
        }
        __syncthreads();

        #pragma unroll
        for (int sm = 0; sm < 2; ++sm) {
            const int row = wm * 32 + sm * 16 + lm;
            V16 a;   // ISA A layout: lane group g holds K {8g..8g+7, 16+8g..+7}
            a.p.lo = *(const v8h*)&sA[cur][row][8 * g];
            a.p.hi = *(const v8h*)&sA[cur][row][16 + 8 * g];
            #pragma unroll
            for (int sn = 0; sn < 2; ++sn) {
                const int col = wn * 32 + sn * 16 + lm;
                V16 b;   // ISA B layout: lane group g holds K {8g..8g+7, 16+8g..+7}
                b.p.lo = *(const v8h*)&sB[cur][col][8 * g];
                b.p.hi = *(const v8h*)&sB[cur][col][16 + 8 * g];
                acc[sm][sn] = wmma_f16(a.v, b.v, acc[sm][sn]);
            }
        }
        __syncthreads();                       // buf cur reused next iter
    }

    const int sec = f0 >> 10;                  // 0=Q 1=K 2=V
    const int h   = (f0 & 1023) >> 6;
    const int b   = m0 >> 11;
    const int t0  = m0 & 2047;

    #pragma unroll
    for (int sm = 0; sm < 2; ++sm)
        #pragma unroll
        for (int sn = 0; sn < 2; ++sn)
            #pragma unroll
            for (int i = 0; i < 8; ++i) {
                int t = t0 + wm * 32 + sm * 16 + 8 * g + i;  // C layout row
                int d = wn * 32 + sn * 16 + lm;              // C layout col
                float v = acc[sm][sn][i];
                size_t head = (size_t)(b * NHEAD + h);
                if (sec == 0)
                    qbuf[(head * SEQ + t) * HD + d] = (_Float16)(v * 0.125f);
                else if (sec == 1)
                    kbuf[(head * SEQ + t) * HD + d] = (_Float16)v;
                else
                    vtbuf[(head * HD + d) * SEQ + t] = (_Float16)v;
            }
}

// ---------------------------------------------------------------------------
// Kernel 2: causal flash attention. Grid (B*H, T/128), 8 waves x 16 rows.
//   K/V^T tiles staged async into double-buffered LDS.
// ---------------------------------------------------------------------------
constexpr int BQ  = 128;
constexpr int BKV = 64;
constexpr int LDK = HD + 8;    // 144B rows

__global__ __launch_bounds__(256) void attn_kernel(
    const _Float16* __restrict__ qbuf, const _Float16* __restrict__ kbuf,
    const _Float16* __restrict__ vtbuf, float* __restrict__ out)
{
    __shared__ _Float16 sK[2][BKV][LDK];       // sK[buf][n][d]
    __shared__ _Float16 sV[2][HD][BKV + 8];    // sV[buf][d][n]
    __shared__ _Float16 sP[8][16][BKV + 8];    // per-wave P staging

    const int bh = blockIdx.x;
    const int q0 = blockIdx.y * BQ;

    const _Float16* Q  = qbuf  + (size_t)bh * SEQ * HD;
    const _Float16* K  = kbuf  + (size_t)bh * SEQ * HD;
    const _Float16* Vt = vtbuf + (size_t)bh * HD * SEQ;
    float*          O  = out   + (size_t)bh * SEQ * HD;

    const int tid  = threadIdx.x;
    const int lane = tid & 31;
    const int wave = tid >> 5;
    const int g    = lane >> 4;
    const int lm   = lane & 15;
    const int r0   = q0 + wave * 16;

    // 4 async b128 ops per thread per kv-tile (sK: 2, sV: 2)
    auto issue_tile = [&](int kb, int buf) {
        const int n0 = kb * BKV;
        #pragma unroll
        for (int j = 0; j < 2; ++j) {
            int idx = tid + j * 256;           // 512 chunks each
            int r   = idx >> 3;
            int c   = (idx & 7) * 8;
            async_b128(&sK[buf][r][c], K  + (size_t)(n0 + r) * HD + c);
            async_b128(&sV[buf][r][c], Vt + (size_t)r * SEQ + n0 + c);
        }
    };

    // this wave's Q tile directly in WMMA A-operand layout (scale pre-folded)
    V16 qa[2];
    {
        const _Float16* qrow = Q + (size_t)(r0 + lm) * HD;
        qa[0].p.lo = *(const v8h*)(qrow + 8 * g);
        qa[0].p.hi = *(const v8h*)(qrow + 16 + 8 * g);
        qa[1].p.lo = *(const v8h*)(qrow + 32 + 8 * g);
        qa[1].p.hi = *(const v8h*)(qrow + 48 + 8 * g);
    }

    v8f  o[4] = {};
    float rm[8], rl[8];
    #pragma unroll
    for (int i = 0; i < 8; ++i) { rm[i] = -1e30f; rl[i] = 0.0f; }

    const int nblocks = q0 / BKV + 2;          // causal: keys up to q0+127

    issue_tile(0, 0);
    for (int kb = 0; kb < nblocks; ++kb) {
        const int n0  = kb * BKV;
        const int cur = kb & 1;
        if (kb + 1 < nblocks) {
            issue_tile(kb + 1, cur ^ 1);
            WAIT_ASYNC(0x4);                   // oldest 4 (= tile kb) retired
        } else {
            WAIT_ASYNC(0x0);
        }
        __syncthreads();

        if (n0 <= r0 + 15) {                   // wave-uniform skip of masked blocks
            // ---- S = Q K^T ----
            v8f s[4] = {};
            #pragma unroll
            for (int kk = 0; kk < 2; ++kk)
                #pragma unroll
                for (int sn = 0; sn < 4; ++sn) {
                    const int col = sn * 16 + lm;
                    V16 b;
                    b.p.lo = *(const v8h*)&sK[cur][col][kk * 32 + 8 * g];
                    b.p.hi = *(const v8h*)&sK[cur][col][kk * 32 + 16 + 8 * g];
                    s[sn] = wmma_f16(qa[kk].v, b.v, s[sn]);
                }

            // ---- causal mask + online softmax (row m = r0 + 8g + i) ----
            float alpha[8];
            #pragma unroll
            for (int i = 0; i < 8; ++i) {
                const int m = r0 + 8 * g + i;
                float mx = -1e30f;
                #pragma unroll
                for (int sn = 0; sn < 4; ++sn) {
                    const int n = n0 + sn * 16 + lm;
                    float v = (n <= m) ? s[sn][i] : -1e30f;
                    s[sn][i] = v;
                    mx = fmaxf(mx, v);
                }
                #pragma unroll
                for (int off = 1; off < 16; off <<= 1)
                    mx = fmaxf(mx, __shfl_xor(mx, off, 32));
                const float mnew = fmaxf(rm[i], mx);
                alpha[i] = __expf(rm[i] - mnew);
                rm[i] = mnew;
                float rs = 0.0f;
                #pragma unroll
                for (int sn = 0; sn < 4; ++sn) {
                    float p = __expf(s[sn][i] - mnew);
                    s[sn][i] = p;
                    rs += p;
                }
                #pragma unroll
                for (int off = 1; off < 16; off <<= 1)
                    rs += __shfl_xor(rs, off, 32);
                rl[i] = rl[i] * alpha[i] + rs;
            }

            #pragma unroll
            for (int dt = 0; dt < 4; ++dt)
                #pragma unroll
                for (int i = 0; i < 8; ++i)
                    o[dt][i] *= alpha[i];

            // ---- P: C-layout -> per-wave LDS -> A-layout ----
            #pragma unroll
            for (int sn = 0; sn < 4; ++sn)
                #pragma unroll
                for (int i = 0; i < 8; ++i)
                    sP[wave][8 * g + i][sn * 16 + lm] = (_Float16)s[sn][i];

            // ---- O += P V ----
            #pragma unroll
            for (int kk = 0; kk < 2; ++kk) {
                V16 pa;
                pa.p.lo = *(const v8h*)&sP[wave][lm][kk * 32 + 8 * g];
                pa.p.hi = *(const v8h*)&sP[wave][lm][kk * 32 + 16 + 8 * g];
                #pragma unroll
                for (int dt = 0; dt < 4; ++dt) {
                    const int drow = dt * 16 + lm;
                    V16 b;
                    b.p.lo = *(const v8h*)&sV[cur][drow][kk * 32 + 8 * g];
                    b.p.hi = *(const v8h*)&sV[cur][drow][kk * 32 + 16 + 8 * g];
                    o[dt] = wmma_f16(pa.v, b.v, o[dt]);
                }
            }
        }
        __syncthreads();                       // buf cur reused next iter
    }

    #pragma unroll
    for (int dt = 0; dt < 4; ++dt)
        #pragma unroll
        for (int i = 0; i < 8; ++i) {
            const int m = r0 + 8 * g + i;
            O[(size_t)m * HD + dt * 16 + lm] = o[dt][i] / rl[i];
        }
}

// ---------------------------------------------------------------------------
extern "C" void kernel_launch(void* const* d_in, const int* in_sizes, int n_in,
                              void* d_out, int out_size, void* d_ws, size_t ws_size,
                              hipStream_t stream) {
    const float* x = (const float*)d_in[0];     // [4,2048,1024]
    const float* w = (const float*)d_in[1];     // [3072,1024]
    float* out = (float*)d_out;                 // [4,16,2048,64]

    const size_t head_elems = (size_t)BATCH * NHEAD * SEQ * HD;  // 8,388,608
    const size_t x_elems    = (size_t)BATCH * SEQ * D_MODEL;     // 8,388,608
    const size_t w_elems    = (size_t)3 * D_MODEL * D_MODEL;     // 3,145,728

    _Float16* qbuf  = (_Float16*)d_ws;
    _Float16* kbuf  = qbuf  + head_elems;
    _Float16* vtbuf = kbuf  + head_elems;
    _Float16* xh    = vtbuf + head_elems;
    _Float16* wh    = xh    + x_elems;          // total ~70 MB workspace

    // Stage 0: one-time f32 -> f16 conversion
    int xn4 = (int)(x_elems / 4), wn4 = (int)(w_elems / 4);
    cvt_f16_kernel<<<(xn4 + 255) / 256, 256, 0, stream>>>(x, xh, xn4);
    cvt_f16_kernel<<<(wn4 + 255) / 256, 256, 0, stream>>>(w, wh, wn4);

    // Stage 1: QKV projection
    dim3 g1((BATCH * SEQ) / GBM, (3 * D_MODEL) / GBN);   // 64 x 48
    qkv_gemm_kernel<<<g1, 256, 0, stream>>>(xh, wh, qbuf, kbuf, vtbuf);

    // Stage 2: causal flash attention
    dim3 g2(BATCH * NHEAD, SEQ / BQ);                    // 64 x 16
    attn_kernel<<<g2, 256, 0, stream>>>(qbuf, kbuf, vtbuf, out);
}
